// VAEDecoder_66760971649073
// MI455X (gfx1250) — matrix operand order repaired and checked by reading, not compiled, not run on hardware
//
#include <hip/hip_runtime.h>

// GGNN message passing for MI455X (gfx1250, wave32, WMMA bf16 path).
// GEMMs use v_wmma_f32_16x16x32_bf16 (fp32 accumulate, inline C=0 on the
// first WMMA of each tile, bias folded into the store epilogue).
// Edge phase is L2-resident float4 gather + f32 atomic scatter-add.

#define N_NODES 50000
#define N_EDGES 800000
#define N_GRAPHS 512
#define DD 128
#define MM 256
#define TT 4
#define GH 256
#define N_PASSES 4
#define ROWB (N_NODES / 16)   // 3125 exact 16-row tiles

typedef __attribute__((ext_vector_type(16))) __bf16 v16bf;
typedef __attribute__((ext_vector_type(8)))  __bf16 v8bf;
typedef __attribute__((ext_vector_type(8)))  float  v8f;

union AFrag { v16bf v; v8bf h[2]; };

__device__ __forceinline__ unsigned short f2bf(float f) {
  unsigned int u = __float_as_uint(f);
  u += 0x7fffu + ((u >> 16) & 1u);          // round-to-nearest-even
  return (unsigned short)(u >> 16);
}
__device__ __forceinline__ float sigm(float x) { return 1.0f / (1.0f + __expf(-x)); }

// A fragment (16x32 bf16) from row-major bf16, lda in elements.
// ISA layout: lane row = lane&15; kbase = (lane>=16)*8; two 8-elem runs at
// K = kbase..kbase+7 and K = kbase+16..kbase+23.
__device__ __forceinline__ v16bf load_a(const unsigned short* A, int row0, int lda,
                                        int k0, int lane) {
  AFrag f;
  const unsigned short* p = A + (size_t)(row0 + (lane & 15)) * lda + k0 + ((lane >> 4) << 3);
  f.h[0] = *(const v8bf*)(p);
  f.h[1] = *(const v8bf*)(p + 16);
  return f.v;
}
// B fragment from pre-packed buffer: 512 bf16 per fragment, lane-major.
__device__ __forceinline__ v16bf load_b(const unsigned short* pk, int frag, int lane) {
  return *(const v16bf*)(pk + ((size_t)frag << 9) + (lane << 4));
}
__device__ __forceinline__ v8f wmma_bf(v16bf a, v16bf b, v8f c) {
  return __builtin_amdgcn_wmma_f32_16x16x32_bf16(false, a, false, b, (short)0, c, false, false);
}
__device__ __forceinline__ v8f vzero8() {
  v8f z = {0.f, 0.f, 0.f, 0.f, 0.f, 0.f, 0.f, 0.f};
  return z;
}

// Repack f32 weight matrix into bf16 WMMA B-fragment order.
// B is (K x NC); isT=1 means B[k][n] = src[n*K + k] (i.e. use src transposed).
__global__ void pack_w(const float* __restrict__ src, unsigned short* __restrict__ dst,
                       int K, int NC, int isT) {
  int tid = blockIdx.x * 256 + threadIdx.x;
  if (tid >= K * NC) return;
  int j = tid & 15, lane = (tid >> 4) & 31, f = tid >> 9;
  int NT = NC >> 4;
  int nt = f % NT, kt = f / NT;
  int k = kt * 32 + ((lane >> 4) << 3) + (j & 7) + ((j >> 3) << 4);   // B layout: kbase=(lane>=16)*8; K = kbase..kbase+7, kbase+16..kbase+23
  int n = nt * 16 + (lane & 15);
  float v = isT ? src[(size_t)n * K + k] : src[(size_t)k * NC + n];
  dst[tid] = f2bf(v);
}

__global__ void embed_k(const int* __restrict__ types, const float* __restrict__ emb,
                        float* __restrict__ h, unsigned short* __restrict__ hbf) {
  int tid = blockIdx.x * 256 + threadIdx.x;
  if (tid >= N_NODES * DD) return;
  int n = tid >> 7, c = tid & 127;
  float v = emb[types[n] * DD + c];
  h[tid] = v;
  hbf[tid] = f2bf(v);
}

// msg = h @ W_msg[t] + b_msg[t] : (50000 x 128) x (128 x 256)
__global__ void __launch_bounds__(128) msg_gemm(const unsigned short* __restrict__ hbf,
                                                const unsigned short* __restrict__ pkW,
                                                const float* __restrict__ bias,
                                                float* __restrict__ out) {
  int lane = threadIdx.x & 31;
  int rb = blockIdx.x * 4 + (threadIdx.x >> 5);
  if (rb >= ROWB) return;                      // wave-uniform: EXEC all-ones for WMMA
  int row0 = rb << 4;
  v16bf a[4];
#pragma unroll
  for (int ks = 0; ks < 4; ++ks) a[ks] = load_a(hbf, row0, DD, ks * 32, lane);
  int colb = lane & 15, rsel = (lane >> 4) << 3;
#pragma unroll
  for (int ct = 0; ct < 16; ++ct) {
    v8f acc = vzero8();                        // first WMMA takes inline C=0
#pragma unroll
    for (int ks = 0; ks < 4; ++ks) acc = wmma_bf(a[ks], load_b(pkW, ks * 16 + ct, lane), acc);
    float b = bias[ct * 16 + colb];
    float* op = out + (size_t)(row0 + rsel) * MM + ct * 16 + colb;
#pragma unroll
    for (int j = 0; j < 8; ++j) op[(size_t)j * MM] = acc[j] + b;
  }
}

// agg[src] += msg[dst] for edges of type t. 4 edges/block, 64 lanes x float4 per edge.
__global__ void __launch_bounds__(256) scatter_k(const int* __restrict__ esrc,
                                                 const int* __restrict__ edst,
                                                 const int* __restrict__ etyp,
                                                 const float* __restrict__ msg,
                                                 float* __restrict__ agg, int t) {
  int e = blockIdx.x * 4 + (threadIdx.x >> 6);
  if (e >= N_EDGES) return;
  if (etyp[e] != t) return;
  int s = esrc[e], d = edst[e];
  int c = (threadIdx.x & 63) << 2;
  const float4 v = *(const float4*)(msg + (size_t)d * MM + c);
  float* ap = agg + (size_t)s * MM + c;
  atomicAdd(ap + 0, v.x);
  atomicAdd(ap + 1, v.y);
  atomicAdd(ap + 2, v.z);
  atomicAdd(ap + 3, v.w);
}

__global__ void relu_cvt(const float* __restrict__ agg, unsigned short* __restrict__ mbf) {
  int tid = blockIdx.x * 256 + threadIdx.x;
  if (tid >= N_NODES * MM) return;
  float v = agg[tid];
  mbf[tid] = f2bf(v > 0.0f ? v : 0.0f);
}

// Fused GRUCell: gi = msgs@W_ih.T + b_ih, gh = h@W_hh.T + b_hh, elementwise update.
// Updates h (f32, in-place) and its bf16 mirror. NT for both packed weights = 384/16 = 24.
__global__ void __launch_bounds__(128) gru_k(const unsigned short* __restrict__ mbf,
                                             unsigned short* __restrict__ hbf,
                                             const unsigned short* __restrict__ pkWih,
                                             const unsigned short* __restrict__ pkWhh,
                                             const float* __restrict__ b_ih,
                                             const float* __restrict__ b_hh,
                                             float* __restrict__ h) {
  int lane = threadIdx.x & 31;
  int rb = blockIdx.x * 4 + (threadIdx.x >> 5);
  if (rb >= ROWB) return;
  int row0 = rb << 4;
  v16bf am[8];
#pragma unroll
  for (int ks = 0; ks < 8; ++ks) am[ks] = load_a(mbf, row0, MM, ks * 32, lane);
  v16bf ah[4];
#pragma unroll
  for (int ks = 0; ks < 4; ++ks) ah[ks] = load_a(hbf, row0, DD, ks * 32, lane);
  int colb = lane & 15, rsel = (lane >> 4) << 3;
#pragma unroll
  for (int ct = 0; ct < 8; ++ct) {
    int c = ct * 16 + colb;
    v8f ir = vzero8(), iz = vzero8(), in_ = vzero8();
#pragma unroll
    for (int ks = 0; ks < 8; ++ks) {
      ir  = wmma_bf(am[ks], load_b(pkWih, ks * 24 + ct, lane), ir);
      iz  = wmma_bf(am[ks], load_b(pkWih, ks * 24 + ct + 8, lane), iz);
      in_ = wmma_bf(am[ks], load_b(pkWih, ks * 24 + ct + 16, lane), in_);
    }
    v8f hr = vzero8(), hz = vzero8(), hn = vzero8();
#pragma unroll
    for (int ks = 0; ks < 4; ++ks) {
      hr = wmma_bf(ah[ks], load_b(pkWhh, ks * 24 + ct, lane), hr);
      hz = wmma_bf(ah[ks], load_b(pkWhh, ks * 24 + ct + 8, lane), hz);
      hn = wmma_bf(ah[ks], load_b(pkWhh, ks * 24 + ct + 16, lane), hn);
    }
    float br = b_ih[c], bz = b_ih[DD + c], bn = b_ih[2 * DD + c];
    float cr = b_hh[c], cz = b_hh[DD + c], cn = b_hh[2 * DD + c];
#pragma unroll
    for (int j = 0; j < 8; ++j) {
      int row = row0 + rsel + j;
      float r = sigm((ir[j] + br) + (hr[j] + cr));
      float z = sigm((iz[j] + bz) + (hz[j] + cz));
      float nn = tanhf((in_[j] + bn) + r * (hn[j] + cn));
      size_t idx = (size_t)row * DD + c;
      float hold = h[idx];
      float hnew = (1.0f - z) * nn + z * hold;
      h[idx] = hnew;
      hbf[idx] = f2bf(hnew);
    }
  }
}

__global__ void attn_k(const float* __restrict__ h, const float* __restrict__ wg,
                       const float* __restrict__ bg, float* __restrict__ attn) {
  int n = blockIdx.x * 256 + threadIdx.x;
  if (n >= N_NODES) return;
  const float4* hp = (const float4*)(h + (size_t)n * DD);
  const float4* wp = (const float4*)wg;
  float s = bg[0];
#pragma unroll
  for (int i = 0; i < 32; ++i) {
    float4 a = hp[i], b = wp[i];
    s += a.x * b.x + a.y * b.y + a.z * b.z + a.w * b.w;
  }
  attn[n] = sigm(s);
}

// h_graph[g] += attn[n] * (h @ W_g + b_g)[n]
__global__ void __launch_bounds__(128) readout_k(const unsigned short* __restrict__ hbf,
                                                 const unsigned short* __restrict__ pkWg,
                                                 const float* __restrict__ b_g,
                                                 const float* __restrict__ attn,
                                                 const int* __restrict__ n2g,
                                                 float* __restrict__ hg) {
  int lane = threadIdx.x & 31;
  int rb = blockIdx.x * 4 + (threadIdx.x >> 5);
  if (rb >= ROWB) return;
  int row0 = rb << 4;
  v16bf a[4];
#pragma unroll
  for (int ks = 0; ks < 4; ++ks) a[ks] = load_a(hbf, row0, DD, ks * 32, lane);
  int colb = lane & 15, rsel = (lane >> 4) << 3;
  float at[8];
  int gg[8];
#pragma unroll
  for (int j = 0; j < 8; ++j) {
    int row = row0 + rsel + j;
    at[j] = attn[row];
    gg[j] = n2g[row];
  }
#pragma unroll
  for (int ct = 0; ct < 16; ++ct) {
    v8f acc = vzero8();
#pragma unroll
    for (int ks = 0; ks < 4; ++ks) acc = wmma_bf(a[ks], load_b(pkWg, ks * 16 + ct, lane), acc);
    float b = b_g[ct * 16 + colb];
#pragma unroll
    for (int j = 0; j < 8; ++j)
      atomicAdd(&hg[(size_t)gg[j] * GH + ct * 16 + colb], at[j] * (acc[j] + b));
  }
}

extern "C" void kernel_launch(void* const* d_in, const int* in_sizes, int n_in,
                              void* d_out, int out_size, void* d_ws, size_t ws_size,
                              hipStream_t stream) {
  (void)in_sizes; (void)n_in; (void)out_size; (void)ws_size;
  const int*   node_types = (const int*)d_in[0];
  const int*   esrc  = (const int*)d_in[1];
  const int*   edst  = (const int*)d_in[2];
  const int*   etyp  = (const int*)d_in[3];
  const int*   n2g   = (const int*)d_in[4];
  const float* emb   = (const float*)d_in[5];
  const float* W_msg = (const float*)d_in[6];
  const float* b_msg = (const float*)d_in[7];
  const float* W_ih  = (const float*)d_in[8];
  const float* W_hh  = (const float*)d_in[9];
  const float* b_ih  = (const float*)d_in[10];
  const float* b_hh  = (const float*)d_in[11];
  const float* wgate = (const float*)d_in[12];
  const float* bgate = (const float*)d_in[13];
  const float* W_g   = (const float*)d_in[14];
  const float* b_g   = (const float*)d_in[15];

  float* h  = (float*)d_out;                          // (N, D)
  float* hg = h + (size_t)N_NODES * DD;               // (512, 256)

  char* ws = (char*)d_ws;
  size_t off = 0;
  auto take = [&](size_t bytes) -> char* {
    char* p = ws + off;
    off = (off + bytes + 255) & ~(size_t)255;
    return p;
  };
  unsigned short* pkWmsg = (unsigned short*)take((size_t)TT * DD * MM * 2);
  unsigned short* pkWih  = (unsigned short*)take((size_t)MM * 3 * DD * 2);
  unsigned short* pkWhh  = (unsigned short*)take((size_t)DD * 3 * DD * 2);
  unsigned short* pkWg   = (unsigned short*)take((size_t)DD * GH * 2);
  unsigned short* hbf    = (unsigned short*)take((size_t)N_NODES * DD * 2);
  unsigned short* mbf    = (unsigned short*)take((size_t)N_NODES * MM * 2);
  float* msg  = (float*)take((size_t)N_NODES * MM * 4);
  float* agg  = (float*)take((size_t)N_NODES * MM * 4);
  float* attn = (float*)take((size_t)N_NODES * 4);

  // Weight repack (deterministic, every call).
  for (int t = 0; t < TT; ++t)
    pack_w<<<(DD * MM + 255) / 256, 256, 0, stream>>>(
        W_msg + (size_t)t * DD * MM, pkWmsg + (size_t)t * DD * MM, DD, MM, 0);
  pack_w<<<(MM * 3 * DD + 255) / 256, 256, 0, stream>>>(W_ih, pkWih, MM, 3 * DD, 1);
  pack_w<<<(DD * 3 * DD + 255) / 256, 256, 0, stream>>>(W_hh, pkWhh, DD, 3 * DD, 1);
  pack_w<<<(DD * GH + 255) / 256, 256, 0, stream>>>(W_g, pkWg, DD, GH, 0);

  embed_k<<<(N_NODES * DD + 255) / 256, 256, 0, stream>>>(node_types, emb, h, hbf);

  dim3 gw((ROWB + 3) / 4), bw(128);
  for (int p = 0; p < N_PASSES; ++p) {
    hipMemsetAsync(agg, 0, (size_t)N_NODES * MM * 4, stream);
    for (int t = 0; t < TT; ++t) {
      msg_gemm<<<gw, bw, 0, stream>>>(hbf, pkWmsg + (size_t)t * DD * MM,
                                      b_msg + (size_t)t * MM, msg);
      scatter_k<<<(N_EDGES + 3) / 4, 256, 0, stream>>>(esrc, edst, etyp, msg, agg, t);
    }
    relu_cvt<<<(N_NODES * MM + 255) / 256, 256, 0, stream>>>(agg, mbf);
    gru_k<<<gw, bw, 0, stream>>>(mbf, hbf, pkWih, pkWhh, b_ih, b_hh, h);
  }

  attn_k<<<(N_NODES + 255) / 256, 256, 0, stream>>>(h, wgate, bgate, attn);
  hipMemsetAsync(hg, 0, (size_t)N_GRAPHS * GH * 4, stream);
  readout_k<<<gw, bw, 0, stream>>>(hbf, pkWg, b_g, attn, n2g, hg);
}
